// AttentionHead_33638183863062
// MI455X (gfx1250) — compile-verified, matrix operand order checked by reading, not run on hardware
//
#include <hip/hip_runtime.h>
#include <hip/hip_bf16.h>

typedef __attribute__((ext_vector_type(16))) _Float16     v16h;
typedef __attribute__((ext_vector_type(8)))  _Float16     v8h;
typedef __attribute__((ext_vector_type(4)))  _Float16     v4h;
typedef __attribute__((ext_vector_type(8)))  float        v8f;
typedef __attribute__((ext_vector_type(4)))  float        f32x4;
typedef __attribute__((ext_vector_type(4)))  unsigned int u32x4;
typedef __attribute__((ext_vector_type(8)))  int          i32x8;
typedef __attribute__((ext_vector_type(4)))  int          i32x4;

#define EMBED 1024
#define SEQ   2048
#define BATCH 4
#define HD    128
#define TILE  128
#define LDW   136   // padded half-stride for 128-wide tiles (272B, 16B multiple)
#define LDX   40    // padded half-stride for 32-wide tiles  (80B,  16B multiple)
#define NW    (EMBED * HD)                 // one weight matrix, elements
#define NX4   ((BATCH * SEQ * EMBED) / 4)  // x element count / 4

// fp16 workspace layout (offsets in halves)
#define QOFF  ((size_t)0)                           // Q  [B][S][D] (pre-scaled)
#define KOFF  ((size_t)BATCH * SEQ * HD)            // K  [B][S][D]
#define VOFF  ((size_t)2 * BATCH * SEQ * HD)        // V^T [B][D][S]
#define XOFF  ((size_t)3 * BATCH * SEQ * HD)        // x_h [B][S][E]
#define WOFF  (XOFF + (size_t)BATCH * SEQ * EMBED)  // W^T_h [3][D n][E k]

// LDS layout for attention kernel (offsets in halves)
#define OFFQ  0
#define OFFP  (TILE * LDW)
#define OFFK0 (2 * TILE * LDW)
#define OFFK1 (3 * TILE * LDW)
#define OFFV0 (4 * TILE * LDW)
#define OFFV1 (5 * TILE * LDW)
#define SMEM2_BYTES (6 * TILE * LDW * 2)   // 208,896 B < 320 KB/WGP

// ---------------------------------------------------------------------------
// 16x32 fp16 A/B fragment load from LDS (CDNA5 16-bit matrix layout:
// lanes 0-15 hold K {0..7,16..23}, lanes 16-31 hold K {8..15,24..31}).
// ---------------------------------------------------------------------------
__device__ inline v16h frag_ld(const _Float16* p, int stride) {
    const int lane = threadIdx.x & 31;
    const int r    = lane & 15;
    const int ko   = (lane >> 4) * 8;
    const _Float16* q = p + r * stride + ko;
    union { v16h v; v8h h[2]; } u;
    u.h[0] = *(const v8h*)(q);        // ds_load_b128
    u.h[1] = *(const v8h*)(q + 16);   // ds_load_b128
    return u.v;
}

// ---------------------------------------------------------------------------
// TDM: async 2D fp16 tile copy global -> LDS, with LDS padding of
// (pad_amount+1) DWORDs every 2^(pad_interval+1) DWORDs.  Tracked on
// TENSORcnt.  D# per CDNA5 ISA 8.3/8.4; 6-arg builtin (clang-23).
// ---------------------------------------------------------------------------
__device__ inline void tdm_load_2d(const _Float16* gsrc, unsigned lds_byte_addr,
                                   unsigned rows, unsigned cols, unsigned stride_elems,
                                   unsigned pad_interval, unsigned pad_amount) {
    unsigned long long ga = (unsigned long long)(size_t)gsrc;
    u32x4 g0;
    g0[0] = 1u;                                        // count=1, user descriptor
    g0[1] = lds_byte_addr;                             // lds_addr
    g0[2] = (unsigned)ga;                              // global_addr[31:0]
    g0[3] = ((unsigned)(ga >> 32) & 0x01FFFFFFu) | (2u << 30);  // addr[56:32] | type=2
    i32x8 g1;
    g1[0] = (int)((1u << 16) | (1u << 20) | (pad_interval << 22) | (pad_amount << 25));
    g1[1] = (int)((cols & 0xFFFFu) << 16);             // tensor_dim0[15:0]
    g1[2] = (int)((cols >> 16) | ((rows & 0xFFFFu) << 16));   // dim0[31:16] | dim1[15:0]
    g1[3] = (int)((rows >> 16) | ((cols & 0xFFFFu) << 16));   // dim1[31:16] | tile_dim0
    g1[4] = (int)(rows & 0xFFFFu);                     // tile_dim1 (tile_dim2 = 0)
    g1[5] = (int)stride_elems;                         // tensor_dim0_stride[31:0]
    g1[6] = 0;
    g1[7] = 0;
    i32x4 z4 = {0, 0, 0, 0};
    i32x8 z8 = {0, 0, 0, 0, 0, 0, 0, 0};
    __builtin_amdgcn_tensor_load_to_lds(g0, g1, z4, z4, z8, 0);
}

// ---------------------------------------------------------------------------
// Kernel 0: one-shot fp16 conversion.  x -> x_h; W -> W^T_h (transposed).
// Wide f32x4 loads; packed stores.  ~53 MB of traffic total.
// ---------------------------------------------------------------------------
__global__ __launch_bounds__(256) void cvt_kernel(
    const float* __restrict__ x,
    const float* __restrict__ wq, const float* __restrict__ wk,
    const float* __restrict__ wv, _Float16* __restrict__ ws) {
    int gid = blockIdx.x * 256 + threadIdx.x;
    if (gid < NX4) {
        f32x4 v = ((const f32x4*)x)[gid];
        v4h h;
#pragma unroll
        for (int k = 0; k < 4; ++k) h[k] = (_Float16)v[k];
        ((v4h*)(ws + XOFF))[gid] = h;
    } else {
        int i = gid - NX4;
        if (i < 3 * NW) {
            int sel = i / NW, rem = i % NW;
            int n = rem & (HD - 1), k = rem >> 7;
            const float* w = sel == 0 ? wq : (sel == 1 ? wk : wv);
            ws[WOFF + (size_t)sel * NW + (size_t)n * EMBED + k] =
                (_Float16)w[(size_t)k * HD + n];
        }
    }
}

// ---------------------------------------------------------------------------
// Kernel 1: QKV projection, fully TDM-fed, double-buffered.
// grid = (B*S/128, 3), 256 threads (8 waves).
// sel 0: Q (pre-scaled) [B][S][D];  sel 1: K [B][S][D];  sel 2: V^T [B][D][S]
// ---------------------------------------------------------------------------
__global__ __launch_bounds__(256) void qkv_proj_kernel(
    const float* __restrict__ bq, const float* __restrict__ bk,
    const float* __restrict__ bv, _Float16* __restrict__ ws) {
    __shared__ _Float16 sX[2][TILE * LDX];   // x tiles [128 s][32 k]
    __shared__ _Float16 sW[2][TILE * LDX];   // W^T tiles [128 n][32 k]
    __shared__ _Float16 sOut[TILE * LDW];    // epilogue bounce buffer

    const int sel     = blockIdx.y;
    const float* bias = sel == 0 ? bq : (sel == 1 ? bk : bv);
    const int batch = blockIdx.x >> 4;
    const int m0    = (blockIdx.x & 15) * TILE;
    const int tid   = threadIdx.x;
    const int wave  = tid >> 5;
    const int lane  = tid & 31;

    const _Float16* xh = ws + XOFF + ((size_t)batch * SEQ + m0) * EMBED;
    const _Float16* wt = ws + WOFF + (size_t)sel * NW;
    const unsigned ldsX = (unsigned)(size_t)(void*)&sX[0][0];
    const unsigned ldsW = (unsigned)(size_t)(void*)&sW[0][0];
    const unsigned bufB = TILE * LDX * 2;   // bytes per buffer

    if (wave == 0) {   // TDM issue ignores EXEC; one wave issues for the block
        tdm_load_2d(xh, ldsX, TILE, 32, EMBED, 3, 3);
        tdm_load_2d(wt, ldsW, TILE, 32, EMBED, 3, 3);
    }

    v8f acc[8] = {};
    for (int c = 0; c < EMBED / 32; ++c) {
        const _Float16* pX = &sX[c & 1][0];
        const _Float16* pW = &sW[c & 1][0];
        if (wave == 0) {
            if (c < EMBED / 32 - 1) {   // prefetch chunk c+1
                int nb = (c + 1) & 1;
                tdm_load_2d(xh + (c + 1) * 32, ldsX + nb * bufB, TILE, 32, EMBED, 3, 3);
                tdm_load_2d(wt + (c + 1) * 32, ldsW + nb * bufB, TILE, 32, EMBED, 3, 3);
                __builtin_amdgcn_s_wait_tensorcnt(2);   // all through chunk c done
            } else {
                __builtin_amdgcn_s_wait_tensorcnt(0);
            }
        }
        __syncthreads();

        v16h a = frag_ld(pX + (wave * 16) * LDX, LDX);
#pragma unroll
        for (int t = 0; t < 8; ++t) {
            v16h b = frag_ld(pW + (t * 16) * LDX, LDX);
            acc[t] = __builtin_amdgcn_wmma_f32_16x16x32_f16(
                false, a, false, b, (short)0, acc[t], false, false);
        }
        __syncthreads();   // reads done before TDM of chunk c+2 overwrites
    }

    // bias (+Q scale) -> fp16 into LDS bounce; transposed for V
    const int hi = lane >> 4, ln = lane & 15;
    const float qscale = 0.088388347648318447f;   // 1/sqrt(128)
#pragma unroll
    for (int t = 0; t < 8; ++t) {
        int n = t * 16 + ln;
        float bb = bias[n];
#pragma unroll
        for (int i = 0; i < 8; ++i) {
            int r = wave * 16 + i + 8 * hi;
            float v = acc[t][i] + bb;
            if (sel == 0) v *= qscale;
            if (sel == 2) sOut[n * LDW + r] = (_Float16)v;
            else          sOut[r * LDW + n] = (_Float16)v;
        }
    }
    __syncthreads();

    // coalesced 16B global stores from the bounce buffer
    if (sel == 2) {
        _Float16* dst = ws + VOFF + (size_t)batch * SEQ * HD;  // [D][S]
        for (int c = 0; c < 8; ++c) {
            int idx = c * 256 + tid;
            int r = idx >> 4, c8 = (idx & 15) * 8;
            *(v8h*)&dst[(size_t)r * SEQ + m0 + c8] = *(const v8h*)&sOut[r * LDW + c8];
        }
    } else {
        _Float16* dst = ws + (sel ? KOFF : QOFF) + (size_t)batch * SEQ * HD; // [S][D]
        for (int c = 0; c < 8; ++c) {
            int idx = c * 256 + tid;
            int r = idx >> 4, c8 = (idx & 15) * 8;
            *(v8h*)&dst[(size_t)(m0 + r) * HD + c8] = *(const v8h*)&sOut[r * LDW + c8];
        }
    }
}

// ---------------------------------------------------------------------------
// Kernel 2: causal flash attention, TDM-fed, K/V double-buffered.
// grid = (S/128, B), 256 threads (8 waves), each wave owns 16 query rows.
// ---------------------------------------------------------------------------
__global__ __launch_bounds__(256) void attn_kernel(
    const _Float16* __restrict__ ws, float* __restrict__ out) {
    extern __shared__ _Float16 smem[];

    const int batch = blockIdx.y;
    const int qt    = blockIdx.x;
    const int q0    = qt * TILE;
    const int tid   = threadIdx.x, wave = tid >> 5, lane = tid & 31;
    const int hi    = lane >> 4,  ln   = lane & 15;
    const size_t SD = (size_t)SEQ * HD;

    const _Float16* Qg = ws + QOFF + (size_t)batch * SD + (size_t)q0 * HD;  // pre-scaled
    const _Float16* Kg = ws + KOFF + (size_t)batch * SD;
    const _Float16* Vg = ws + VOFF + (size_t)batch * SD;                    // [D][S]
    const unsigned lds0 = (unsigned)(size_t)(void*)smem;

    if (wave == 0) {
        tdm_load_2d(Qg, lds0 + 2 * OFFQ,  TILE, HD, HD, 5, 3);
        tdm_load_2d(Kg, lds0 + 2 * OFFK0, TILE, HD, HD, 5, 3);
        tdm_load_2d(Vg, lds0 + 2 * OFFV0, HD, TILE, SEQ, 5, 3);
    }

    v8f o_acc[8] = {};
    float m_i[8], l_i[8];
#pragma unroll
    for (int i = 0; i < 8; ++i) { m_i[i] = -3.0e38f; l_i[i] = 0.f; }

    for (int j = 0; j <= qt; ++j) {
        const _Float16* sK = smem + ((j & 1) ? OFFK1 : OFFK0);
        const _Float16* sV = smem + ((j & 1) ? OFFV1 : OFFV0);

        if (wave == 0) {
            if (j < qt) {   // prefetch tile j+1 into the other buffer
                int nb = (j + 1) & 1;
                tdm_load_2d(Kg + (size_t)(j + 1) * TILE * HD,
                            lds0 + 2 * (nb ? OFFK1 : OFFK0), TILE, HD, HD, 5, 3);
                tdm_load_2d(Vg + (size_t)(j + 1) * TILE,
                            lds0 + 2 * (nb ? OFFV1 : OFFV0), HD, TILE, SEQ, 5, 3);
                __builtin_amdgcn_s_wait_tensorcnt(2);  // all through tile j done
            } else {
                __builtin_amdgcn_s_wait_tensorcnt(0);
            }
        }
        __syncthreads();

        // S = Q K^T : wave's 16 rows x 128 kv-cols, K-dim 128 = 4 x 32
        v8f s[8];
#pragma unroll
        for (int t = 0; t < 8; ++t) {
            v8f c0 = {};
#pragma unroll
            for (int kc = 0; kc < 4; ++kc) {
                v16h a = frag_ld(&smem[OFFQ + (wave * 16) * LDW + kc * 32], LDW);
                v16h b = frag_ld(&sK[(t * 16) * LDW + kc * 32], LDW);
                c0 = __builtin_amdgcn_wmma_f32_16x16x32_f16(
                    false, a, false, b, (short)0, c0, false, false);
            }
            s[t] = c0;
        }

        if (j == qt) {   // diagonal tile: causal mask
#pragma unroll
            for (int t = 0; t < 8; ++t)
#pragma unroll
                for (int i = 0; i < 8; ++i) {
                    int rg = q0 + wave * 16 + i + 8 * hi;
                    int cg = j * TILE + t * 16 + ln;
                    if (cg > rg) s[t][i] = -3.0e38f;
                }
        }

        // online softmax; P (fp16) into private sP rows — no barrier needed
#pragma unroll
        for (int i = 0; i < 8; ++i) {
            float rmax = -3.0e38f;
#pragma unroll
            for (int t = 0; t < 8; ++t) rmax = fmaxf(rmax, s[t][i]);
#pragma unroll
            for (int off = 8; off >= 1; off >>= 1)
                rmax = fmaxf(rmax, __shfl_xor(rmax, off, 32));
            float mnew  = fmaxf(m_i[i], rmax);
            float alpha = __expf(m_i[i] - mnew);
            float rsum  = 0.f;
            int   r     = wave * 16 + i + 8 * hi;
#pragma unroll
            for (int t = 0; t < 8; ++t) {
                float p = __expf(s[t][i] - mnew);
                rsum += p;
                smem[OFFP + r * LDW + t * 16 + ln] = (_Float16)p;
            }
#pragma unroll
            for (int off = 8; off >= 1; off >>= 1)
                rsum += __shfl_xor(rsum, off, 32);
            l_i[i] = l_i[i] * alpha + rsum;
            m_i[i] = mnew;
#pragma unroll
            for (int t = 0; t < 8; ++t) o_acc[t][i] *= alpha;
        }

        // O += P V^T-tile : reduce over n = 128 in chunks of 32
#pragma unroll
        for (int dt = 0; dt < 8; ++dt) {
#pragma unroll
            for (int nc = 0; nc < 4; ++nc) {
                v16h a = frag_ld(&smem[OFFP + (wave * 16) * LDW + nc * 32], LDW);
                v16h b = frag_ld(&sV[(dt * 16) * LDW + nc * 32], LDW);
                o_acc[dt] = __builtin_amdgcn_wmma_f32_16x16x32_f16(
                    false, a, false, b, (short)0, o_acc[dt], false, false);
            }
        }
        __syncthreads();   // compute done before next iteration's TDM issue
    }

    float* outp = out + ((size_t)batch * SEQ) * HD;
#pragma unroll
    for (int i = 0; i < 8; ++i) {
        float inv = 1.0f / l_i[i];
        int   r   = q0 + wave * 16 + i + 8 * hi;
#pragma unroll
        for (int dt = 0; dt < 8; ++dt)
            outp[(size_t)r * HD + dt * 16 + ln] = o_acc[dt][i] * inv;
    }
}

extern "C" void kernel_launch(void* const* d_in, const int* in_sizes, int n_in,
                              void* d_out, int out_size, void* d_ws, size_t ws_size,
                              hipStream_t stream) {
    const float* x  = (const float*)d_in[0];
    const float* wq = (const float*)d_in[1];
    const float* bq = (const float*)d_in[2];
    const float* wk = (const float*)d_in[3];
    const float* bk = (const float*)d_in[4];
    const float* wv = (const float*)d_in[5];
    const float* bv = (const float*)d_in[6];
    _Float16* ws = (_Float16*)d_ws;   // fp16 x_h, W^T_h, Q, K, V^T: ~24 MB

    int cvt_items  = NX4 + 3 * NW;
    int cvt_blocks = (cvt_items + 255) / 256;
    cvt_kernel<<<cvt_blocks, 256, 0, stream>>>(x, wq, wk, wv, ws);
    qkv_proj_kernel<<<dim3(BATCH * SEQ / TILE, 3), 256, 0, stream>>>(
        bq, bk, bv, ws);
    attn_kernel<<<dim3(SEQ / TILE, BATCH), 256, SMEM2_BYTES, stream>>>(
        ws, (float*)d_out);
}
